// TemEmbedding_4535485464585
// MI455X (gfx1250) — matrix operand order repaired and checked by reading, not compiled, not run on hardware
//
#include <hip/hip_runtime.h>
#include <hip/hip_bf16.h>

typedef __attribute__((ext_vector_type(16))) _Float16 v16h;
typedef __attribute__((ext_vector_type(8)))  _Float16 v8h;
typedef __attribute__((ext_vector_type(8)))  float    v8f;

static constexpr int kD      = 128;
static constexpr int kNumDow = 7;
static constexpr int kNumTod = 288;
static constexpr int kNRows  = kNumDow * kNumTod;   // 2016

// ---------------------------------------------------------------------------
// Kernel 1: precompute hTable[dow*288+tod][d] = f16(relu(W1[dow]+W1[7+tod]+b1))
// 2016 rows x 128 f16 = 516 KB, L2-resident on a 192MB-L2 part.
// ---------------------------------------------------------------------------
__global__ void build_htable_kernel(const float* __restrict__ W1,
                                    const float* __restrict__ b1,
                                    _Float16* __restrict__ hTable) {
    const int row = blockIdx.x;          // 0..2015
    const int d   = threadIdx.x;         // 0..127
    const int dow = row / kNumTod;
    const int tod = row - dow * kNumTod;
    float v = W1[dow * kD + d] + W1[(kNumDow + tod) * kD + d] + b1[d];
    v = fmaxf(v, 0.0f);
    hTable[row * kD + d] = (_Float16)v;
}

// ---------------------------------------------------------------------------
// Kernel 2: convert W2 (128x128, row-major [K][N]) to f16
// ---------------------------------------------------------------------------
__global__ void cvt_w2_kernel(const float* __restrict__ W2,
                              _Float16* __restrict__ W2h) {
    const int i = blockIdx.x * blockDim.x + threadIdx.x;
    W2h[i] = (_Float16)W2[i];
}

// ---------------------------------------------------------------------------
// Main kernel: 8 waves / block; wave w owns output columns [16w, 16w+16).
// Each block processes 256 tokens = 16 tiles of 16 tokens.
// Per tile: 4 x v_wmma_f32_16x16x32_f16 accumulate K=128.
// ---------------------------------------------------------------------------
__global__ __launch_bounds__(256) void te_ffn_wmma_kernel(
    const unsigned* __restrict__ te32,     // TE int64 viewed as u32 dwords
    const _Float16* __restrict__ hTable,
    const _Float16* __restrict__ W2h,
    const float*    __restrict__ b2,
    float*          __restrict__ out) {

    const int  lane   = threadIdx.x & 31;
    const int  wave   = threadIdx.x >> 5;        // 0..7
    const int  n0     = wave * 16;               // output column tile base
    const int  col    = n0 + (lane & 15);
    const bool hiHalf = (lane >= 16);

    // ---- B fragments (loop-invariant): V_j holds K-pair (2j,2j+1); lanes=N;
    //      lanes 16-31 hold K+16.  4 fragments cover K=0..127.
    v16h bf[4];
#pragma unroll
    for (int kt = 0; kt < 4; ++kt) {
        const int kb = kt * 32 + (hiHalf ? 16 : 0);
#pragma unroll
        for (int j = 0; j < 8; ++j) {
            bf[kt][2 * j]     = W2h[(kb + 2 * j)     * kD + col];
            bf[kt][2 * j + 1] = W2h[(kb + 2 * j + 1) * kD + col];
        }
    }
    const float bias = b2[col];

    const int blockTokBase = blockIdx.x * 256;

#pragma unroll 1
    for (int t = 0; t < 16; ++t) {
        const int tokBase = blockTokBase + t * 16;
        const int tok     = tokBase + (lane & 15);   // row M = lane&15

        // TE[tok][0], TE[tok][1] are int64; values < 288 so low dwords suffice.
        const unsigned t0 = te32[(size_t)tok * 4 + 0];
        const unsigned t1 = te32[(size_t)tok * 4 + 2];
        const unsigned row = (t0 % kNumDow) * kNumTod + (t1 % kNumTod);
        const _Float16* rowp = hTable + (size_t)row * kD + (hiHalf ? 16 : 0);

        v8f c = {};
#pragma unroll
        for (int kt = 0; kt < 4; ++kt) {
            // A in contiguous k order, same as B: lanes 0-15 hold K{0..15}, lanes 16-31
            // hold K{16..31} -> two aligned 16B loads per K-step.
            v8h lo = *(const v8h*)(rowp + kt * 32);
            v8h hi = *(const v8h*)(rowp + kt * 32 + 8);
            v16h a = __builtin_shufflevector(lo, hi,
                0, 1, 2, 3, 4, 5, 6, 7, 8, 9, 10, 11, 12, 13, 14, 15);
            c = __builtin_amdgcn_wmma_f32_16x16x32_f16(
                    /*neg_a=*/false, a, /*neg_b=*/false, bf[kt],
                    /*c_mod=*/(short)0, c, /*reuse_a=*/false, /*reuse_b=*/false);
        }

        // C/D layout: VGPR r -> M=r (lanes 0-15) or M=8+r (lanes 16-31); N=lane&15
        float* orow = out + (size_t)tokBase * kD + col;
        const int mb = hiHalf ? 8 : 0;
#pragma unroll
        for (int r = 0; r < 8; ++r) {
            orow[(size_t)(mb + r) * kD] = c[r] + bias;
        }
    }
}

// ---------------------------------------------------------------------------
// Launch
// ---------------------------------------------------------------------------
extern "C" void kernel_launch(void* const* d_in, const int* in_sizes, int n_in,
                              void* d_out, int out_size, void* d_ws, size_t ws_size,
                              hipStream_t stream) {
    // setup_inputs order: TE(int64), T(scalar, unused), W1, b1, W2, b2
    const unsigned* te32 = (const unsigned*)d_in[0];
    const float*    W1   = (const float*)d_in[2];
    const float*    b1   = (const float*)d_in[3];
    const float*    W2   = (const float*)d_in[4];
    const float*    b2   = (const float*)d_in[5];
    float*          out  = (float*)d_out;

    // Workspace layout: hTable (516096 B) | W2h (32768 B)
    _Float16* hTable = (_Float16*)d_ws;
    _Float16* W2h    = (_Float16*)((char*)d_ws + (size_t)kNRows * kD * sizeof(_Float16));

    build_htable_kernel<<<kNRows, kD, 0, stream>>>(W1, b1, hTable);
    cvt_w2_kernel<<<(kD * kD) / 256, 256, 0, stream>>>(W2, W2h);

    const int ntok = in_sizes[0] / 2;            // B*S = 131072
    te_ffn_wmma_kernel<<<ntok / 256, 256, 0, stream>>>(te32, hTable, W2h, b2, out);
}
